// ContrastiveLoss_25494925869443
// MI455X (gfx1250) — matrix operand order repaired and checked by reading, compiled, not run on hardware
//
#include <hip/hip_runtime.h>
#include <hip/hip_bf16.h>
#include <math.h>

// ---------------- problem constants ----------------
#define B_        256
#define TWOB      512
#define DIM       2048
#define HW        64
#define TEMP_INV  10.0f
#define BN_EPS    1e-5f
#define COS_EPS   1e-8f
#define SLICE_RNG 2

typedef __attribute__((ext_vector_type(8)))  float        v8f;
typedef __attribute__((ext_vector_type(16))) __bf16       v16bf;
typedef __attribute__((ext_vector_type(4)))  unsigned int v4u;
typedef __attribute__((ext_vector_type(8)))  int          v8i;
typedef __attribute__((ext_vector_type(4)))  int          v4i;

union Frag { uint4 u[2]; v16bf v; };   // 32 bytes = 16 bf16 per lane

__device__ __forceinline__ unsigned short f2bf(float f) {
    unsigned u = __float_as_uint(f);
    unsigned r = 0x7FFFu + ((u >> 16) & 1u);     // round-to-nearest-even
    return (unsigned short)((u + r) >> 16);
}
__device__ __forceinline__ float bf2f(unsigned short h) {
    return __uint_as_float(((unsigned)h) << 16);
}

// ---- TDM: stage one contiguous panel (bytes % 16 == 0) from global into LDS.
// Issued by wave 0 only (TDM is a whole-wave op, EXEC ignored, TENSORcnt-tracked).
// Caller must __syncthreads() after. Descriptor per CDNA5 ISA ch.8:
//   group0: [1:0]=count=1 | lds_addr | global_addr[56:0] | type=2
//   group1: data_size=3 (8B), tensor_dim0 = tile_dim0 = n8, dim1=1, stride0=n8
// This toolchain exposes the 6-arg builtin: (v4u, v8i, v4i, v4i, v8i, i32 cpol).
__device__ __forceinline__ void stage_panel(unsigned short* ldsDst,
                                            const unsigned short* gsrc,
                                            unsigned bytes, int tid) {
#if __has_builtin(__builtin_amdgcn_tensor_load_to_lds) && __has_builtin(__builtin_amdgcn_s_wait_tensorcnt)
    if (tid < 32) {
        unsigned n8 = bytes >> 3;                               // 8-byte elements
        unsigned long long ga = (unsigned long long)(uintptr_t)gsrc;
        unsigned lds_off = (unsigned)(uintptr_t)ldsDst;         // low 32b of flat = LDS byte offset
        v4u g0 = { 1u,                                          // count=1, user D#, no gather
                   lds_off,                                     // lds_addr (bits 63:32)
                   (unsigned)ga,                                // global_addr[31:0]
                   ((unsigned)(ga >> 32) & 0x01FFFFFFu) | (2u << 30) }; // addr[56:32] | type=2
        v8i g1 = { (int)(3u << 16),                             // data_size=8B, wg_mask=0
                   (int)((n8 & 0xFFFFu) << 16),                 // tensor_dim0[15:0]
                   (int)((n8 >> 16) | (1u << 16)),              // tensor_dim0[31:16] | tensor_dim1=1
                   (int)((n8 & 0xFFFFu) << 16),                 // tensor_dim1[31:16]=0 | tile_dim0=n8
                   1,                                           // tile_dim1=1, tile_dim2=0
                   (int)n8, 0, 0 };                             // tensor_dim0_stride=n8
        v4i gz4 = { 0, 0, 0, 0 };
        v8i gz8 = { 0, 0, 0, 0, 0, 0, 0, 0 };
        __builtin_amdgcn_tensor_load_to_lds(g0, g1, gz4, gz4, gz8, 0);
        __builtin_amdgcn_s_wait_tensorcnt(0);
    }
#else
    // cooperative fallback: plain coalesced copy
    const uint4* s = (const uint4*)gsrc;
    uint4*       d = (uint4*)ldsDst;
    for (unsigned i = tid; i < (bytes >> 4); i += 256) d[i] = s[i];
#endif
}

// ---------------- 0: zero accumulators + outputs ----------------
__global__ void cl_init_kernel(float* out, float* colsum, float* colsumsq, float* rowsq) {
    int i = blockIdx.x * blockDim.x + threadIdx.x;
    if (i < 2)    out[i]      = 0.0f;
    if (i < DIM)  { colsum[i] = 0.0f; colsumsq[i] = 0.0f; }
    if (i < TWOB) rowsq[i]    = 0.0f;
}

// ---------------- 1: convert W1/W2 to bf16 ----------------
__global__ void cl_cvtw_kernel(const float* __restrict__ W1, const float* __restrict__ W2,
                               unsigned short* __restrict__ w1b, unsigned short* __restrict__ w2b) {
    int i = blockIdx.x * blockDim.x + threadIdx.x;
    if (i < DIM * DIM) { w1b[i] = f2bf(W1[i]); w2b[i] = f2bf(W2[i]); }
}

// ---------------- 2: spatial mean pool ----------------
// Half-wave (16 lanes x float4 = 64 floats) per (b,d) chunk: each wave issues one
// fully coalesced 512B load covering two chunks; 4-step xor reduction per half.
__global__ void cl_pool_kernel(const float* __restrict__ z0, const float* __restrict__ z1,
                               unsigned short* __restrict__ pbf) {
    int wave = (blockIdx.x * blockDim.x + threadIdx.x) >> 5;
    int lane = threadIdx.x & 31;
    int sub  = lane & 15, half = lane >> 4;
    int chunk = wave * 2 + half;
    int b = chunk >> 11;              // / DIM
    int d = chunk & (DIM - 1);
    const float* src = (b < B_)
        ? z0 + ((size_t)b * DIM + d) * HW
        : z1 + ((size_t)(b - B_) * DIM + d) * HW;
    float4 v = *(const float4*)(src + sub * 4);
    float  s = (v.x + v.y) + (v.z + v.w);
#pragma unroll
    for (int m = 8; m >= 1; m >>= 1) s += __shfl_xor(s, m, 32);   // stays within 16-lane half
    if (sub == 0) pbf[chunk] = f2bf(s * (1.0f / HW));
}

// ---------------- WMMA 16x16 tile: A panel in LDS, B row-major in global -----
// Computes A @ B^T. ISA 7.12.2 fragment layouts:
//  A lane: row = (lane&15); VGPR0-3 = K[hi*8 .. +7], VGPR4-7 = K[16+hi*8 .. +7]
//  B lane: col = colBase+(lane&15); VGPR0-3 = K[hi*8 .. +7], VGPR4-7 = K[16+hi*8 .. +7]
__device__ __forceinline__ v8f wmma_abt_lds(const unsigned short* __restrict__ aP,
                                            const unsigned short* __restrict__ Bm,
                                            int colBase, int K) {
    int lane = threadIdx.x & 31;
    int sub  = lane & 15, hi = lane >> 4;
    const unsigned short* ar = aP + (size_t)sub * K;                 // LDS -> ds_load_b128
    const unsigned short* br = Bm + (size_t)(colBase + sub) * K;     // global, L2-resident
    v8f acc = {};
    for (int k = 0; k < K; k += 32) {
        Frag a, b;
        a.u[0] = *(const uint4*)(ar + k + hi * 8);
        a.u[1] = *(const uint4*)(ar + k + 16 + hi * 8);
        b.u[0] = *(const uint4*)(br + k + hi * 8);
        b.u[1] = *(const uint4*)(br + k + 16 + hi * 8);
        acc = __builtin_amdgcn_wmma_f32_16x16x32_bf16(false, a.v, false, b.v,
                                                      (short)0, acc, false, false);
    }
    return acc;
}

// ---------------- 3: h = p @ W1^T ; fused BN column sum / sumsq --------------
// 8 waves/block share one 16-row A panel (64KB) staged once via TDM.
__global__ void cl_gemm1_kernel(const unsigned short* __restrict__ pbf,
                                const unsigned short* __restrict__ w1b,
                                float* __restrict__ h,
                                float* __restrict__ colsum, float* __restrict__ colsumsq) {
    extern __shared__ unsigned short smem[];                   // 16*DIM bf16 = 64KB
    int tid = threadIdx.x, widx = tid >> 5, lane = tid & 31;
    int rowBase = (blockIdx.x >> 4) << 4;                      // same for all 8 waves
    int colBase = (((blockIdx.x & 15) << 3) + widx) << 4;
    stage_panel(smem, pbf + (size_t)rowBase * DIM, 16 * DIM * 2, tid);
    __syncthreads();
    v8f acc = wmma_abt_lds(smem, w1b, colBase, DIM);
    int sub = lane & 15, hi = lane >> 4;
    int col = colBase + sub;
    float s = 0.0f, q = 0.0f;
#pragma unroll
    for (int r = 0; r < 8; ++r) {                              // D: VGPR r -> row r+hi*8, col sub
        float v = acc[r];
        h[(size_t)(rowBase + r + hi * 8) * DIM + col] = v;
        s += v; q += v * v;
    }
    s += __shfl_xor(s, 16, 32);                                // combine the two 8-row halves
    q += __shfl_xor(q, 16, 32);
    if (hi == 0) { atomicAdd(&colsum[col], s); atomicAdd(&colsumsq[col], q); }
}

// ---------------- 4: batchnorm (b1 cancels through BN) + relu -> bf16 --------
__global__ void cl_bnrelu_kernel(const float* __restrict__ h,
                                 const float* __restrict__ colsum, const float* __restrict__ colsumsq,
                                 const float* __restrict__ gamma, const float* __restrict__ beta,
                                 unsigned short* __restrict__ hnb) {
    int i = blockIdx.x * blockDim.x + threadIdx.x;
    int j = i & (DIM - 1);
    float mu  = colsum[j]   * (1.0f / TWOB);
    float var = colsumsq[j] * (1.0f / TWOB) - mu * mu;
    float v = (h[i] - mu) * rsqrtf(var + BN_EPS) * gamma[j] + beta[j];
    hnb[i] = f2bf(fmaxf(v, 0.0f));
}

// ---------------- 5: z = hn @ W2^T + b2 ; fused row sum-of-squares -----------
__global__ void cl_gemm2_kernel(const unsigned short* __restrict__ hnb,
                                const unsigned short* __restrict__ w2b,
                                const float* __restrict__ b2,
                                float* __restrict__ z, float* __restrict__ rowsq) {
    extern __shared__ unsigned short smem[];
    int tid = threadIdx.x, widx = tid >> 5, lane = tid & 31;
    int rowBase = (blockIdx.x >> 4) << 4;
    int colBase = (((blockIdx.x & 15) << 3) + widx) << 4;
    stage_panel(smem, hnb + (size_t)rowBase * DIM, 16 * DIM * 2, tid);
    __syncthreads();
    v8f acc = wmma_abt_lds(smem, w2b, colBase, DIM);
    int sub = lane & 15, hi = lane >> 4;
    int col  = colBase + sub;
    float bias = b2[col];
#pragma unroll
    for (int r = 0; r < 8; ++r) {
        float v = acc[r] + bias;
        int row = rowBase + r + hi * 8;
        z[(size_t)row * DIM + col] = v;
        float q = v * v;                  // reduce the 16 columns of this row
        q += __shfl_xor(q, 1, 32); q += __shfl_xor(q, 2, 32);
        q += __shfl_xor(q, 4, 32); q += __shfl_xor(q, 8, 32);
        if (sub == 0) atomicAdd(&rowsq[row], q);
    }
}

// ---------------- 6: zn = z / max(||z||, eps) -> split bf16 (hi + residual) --
__global__ void cl_znorm_kernel(const float* __restrict__ z, const float* __restrict__ rowsq,
                                unsigned short* __restrict__ zhib,
                                unsigned short* __restrict__ zlob) {
    int i = blockIdx.x * blockDim.x + threadIdx.x;
    int row = i >> 11;                    // / DIM
    float n = fmaxf(sqrtf(rowsq[row]), COS_EPS);
    float x = z[i] / n;
    unsigned short hs = f2bf(x);
    zhib[i] = hs;
    zlob[i] = f2bf(x - bf2f(hs));         // residual: recovers ~fp32 in hi*hi + hi*lo + lo*hi
}

// ---------------- 7: sim = zn @ zn^T, split-bf16 (3 WMMAs / k-step) ----------
__global__ void cl_gemm3_kernel(const unsigned short* __restrict__ zhib,
                                const unsigned short* __restrict__ zlob,
                                float* __restrict__ sim) {
    extern __shared__ unsigned short smem[];                   // hi panel | lo panel = 128KB
    int tid = threadIdx.x, widx = tid >> 5, lane = tid & 31;
    int rowBase = (blockIdx.x >> 2) << 4;
    int colBase = (((blockIdx.x & 3) << 3) + widx) << 4;
    unsigned short* aHi = smem;
    unsigned short* aLo = smem + 16 * DIM;
    stage_panel(aHi, zhib + (size_t)rowBase * DIM, 16 * DIM * 2, tid);
    stage_panel(aLo, zlob + (size_t)rowBase * DIM, 16 * DIM * 2, tid);
    __syncthreads();
    int sub = lane & 15, hi = lane >> 4;
    const unsigned short* arh = aHi + (size_t)sub * DIM;
    const unsigned short* arl = aLo + (size_t)sub * DIM;
    const unsigned short* brh = zhib + (size_t)(colBase + sub) * DIM;
    const unsigned short* brl = zlob + (size_t)(colBase + sub) * DIM;
    v8f acc = {};
    for (int k = 0; k < DIM; k += 32) {
        Frag ah, al, bh, bl;
        ah.u[0] = *(const uint4*)(arh + k + hi * 8);
        ah.u[1] = *(const uint4*)(arh + k + 16 + hi * 8);
        al.u[0] = *(const uint4*)(arl + k + hi * 8);
        al.u[1] = *(const uint4*)(arl + k + 16 + hi * 8);
        bh.u[0] = *(const uint4*)(brh + k + hi * 8);
        bh.u[1] = *(const uint4*)(brh + k + 16 + hi * 8);
        bl.u[0] = *(const uint4*)(brl + k + hi * 8);
        bl.u[1] = *(const uint4*)(brl + k + 16 + hi * 8);
        acc = __builtin_amdgcn_wmma_f32_16x16x32_bf16(false, ah.v, false, bh.v, (short)0, acc, false, false);
        acc = __builtin_amdgcn_wmma_f32_16x16x32_bf16(false, ah.v, false, bl.v, (short)0, acc, false, false);
        acc = __builtin_amdgcn_wmma_f32_16x16x32_bf16(false, al.v, false, bh.v, (short)0, acc, false, false);
    }
#pragma unroll
    for (int r = 0; r < 8; ++r)
        sim[(size_t)(rowBase + r + hi * 8) * TWOB + colBase + sub] = acc[r];
}

// ---------------- 8: InfoNCE-style loss, one block per anchor row ------------
__global__ void cl_loss_kernel(const float* __restrict__ sim, const int* __restrict__ rel,
                               float* __restrict__ out) {
    __shared__ float red[256];
    int i = blockIdx.x, t = threadIdx.x;
    float S1 = expf(sim[(size_t)i * TWOB + t]      * TEMP_INV);   // S[i, t]      (t < B)
    float S2 = expf(sim[(size_t)i * TWOB + B_ + t] * TEMP_INV);   // S[i, B + t]
    int  d   = rel[i] - rel[t]; if (d < 0) d = -d;
    bool pos = (d <= SLICE_RNG) && (t != i);
    bool neg = (d > SLICE_RNG);

    red[t] = (neg ? S1 : 0.0f) + S2;                 // negative-pool sum
    __syncthreads();
    for (int o = 128; o >= 1; o >>= 1) { if (t < o) red[t] += red[t + o]; __syncthreads(); }
    float neg_sum = red[0];
    __syncthreads();

    red[t] = pos ? log1pf(neg_sum / S1) : 0.0f;      // -log(S/(S+neg))
    __syncthreads();
    for (int o = 128; o >= 1; o >>= 1) { if (t < o) red[t] += red[t + o]; __syncthreads(); }
    float tsum = red[0];
    __syncthreads();

    red[t] = pos ? 1.0f : 0.0f;                      // positive count
    __syncthreads();
    for (int o = 128; o >= 1; o >>= 1) { if (t < o) red[t] += red[t + o]; __syncthreads(); }
    float cnt = red[0];

    if (t == 0 && cnt > 0.5f) {
        atomicAdd(&out[0], tsum / cnt);              // loss
        atomicAdd(&out[1], 1.0f);                    // n_defined
    }
}

// ---------------- host side ----------------
extern "C" void kernel_launch(void* const* d_in, const int* in_sizes, int n_in,
                              void* d_out, int out_size, void* d_ws, size_t ws_size,
                              hipStream_t stream) {
    (void)in_sizes; (void)n_in; (void)out_size; (void)ws_size;
    // setup_inputs order: z0, z1, rel0, rel1, W1, b1, gamma, beta, W2, b2
    const float* z0    = (const float*)d_in[0];
    const float* z1    = (const float*)d_in[1];
    const int*   rel0  = (const int*)  d_in[2];
    const float* W1    = (const float*)d_in[4];    // b1 (d_in[5]) cancels through BN
    const float* gamma = (const float*)d_in[6];
    const float* beta  = (const float*)d_in[7];
    const float* W2    = (const float*)d_in[8];
    const float* b2    = (const float*)d_in[9];
    float* out = (float*)d_out;

    char* ws = (char*)d_ws;
    size_t off = 0;
    auto take = [&](size_t bytes) { char* p = ws + off; off += (bytes + 255) & ~(size_t)255; return p; };
    unsigned short* pbf      = (unsigned short*)take((size_t)TWOB * DIM * 2);
    unsigned short* w1b      = (unsigned short*)take((size_t)DIM * DIM * 2);
    unsigned short* w2b      = (unsigned short*)take((size_t)DIM * DIM * 2);
    float*          h        = (float*)         take((size_t)TWOB * DIM * 4);
    unsigned short* hnb      = (unsigned short*)take((size_t)TWOB * DIM * 2);
    float*          sim      = (float*)         take((size_t)TWOB * TWOB * 4);
    float*          colsum   = (float*)         take(DIM * 4);
    float*          colsumsq = (float*)         take(DIM * 4);
    float*          rowsq    = (float*)         take(TWOB * 4);
    float*          z        = h;                 // h is dead after bn_relu
    unsigned short* zhib     = pbf;               // p is dead after gemm1
    unsigned short* zlob     = hnb;               // hn is dead after gemm2

    cl_init_kernel  <<<8, 256, 0, stream>>>(out, colsum, colsumsq, rowsq);
    cl_cvtw_kernel  <<<(DIM * DIM) / 256, 256, 0, stream>>>(W1, W2, w1b, w2b);
    cl_pool_kernel  <<<(TWOB * DIM) / 16, 256, 0, stream>>>(z0, z1, pbf);    // 2 chunks / wave
    cl_gemm1_kernel <<<512, 256, 16 * DIM * 2, stream>>>(pbf, w1b, h, colsum, colsumsq);
    cl_bnrelu_kernel<<<(TWOB * DIM) / 256, 256, 0, stream>>>(h, colsum, colsumsq, gamma, beta, hnb);
    cl_gemm2_kernel <<<512, 256, 16 * DIM * 2, stream>>>(hnb, w2b, b2, z, rowsq);
    cl_znorm_kernel <<<(TWOB * DIM) / 256, 256, 0, stream>>>(z, rowsq, zhib, zlob);
    cl_gemm3_kernel <<<128, 256, 2 * 16 * DIM * 2, stream>>>(zhib, zlob, sim);
    cl_loss_kernel  <<<B_, 256, 0, stream>>>(sim, rel0, out);
}
